// LUPVisQNet_29377576305090
// MI455X (gfx1250) — compile-verified, matrix operand order checked by reading, not run on hardware
//
#include <hip/hip_runtime.h>
#include <hip/hip_bf16.h>

#define TAU_INV 10.0f
#define BB 128
#define F_IN 1000
#define KPAD1 1024
#define DD 512
#define CC 5
#define KK 10
#define TT 1000

typedef __attribute__((ext_vector_type(16))) _Float16 v16h;
typedef __attribute__((ext_vector_type(8)))  float    v8f;

// K offset of dword j (pair of f16) inside a 16-bit A/B fragment,
// per ISA 7.12.2 "16-bit A-Matrix 16x32": VGPR0..3 = K 0..7 (lo-half lanes)
// / 8..15 (hi-half lanes); VGPR4..7 = K 16..23 / 24..31.
__device__ __forceinline__ int frag_kbase(int j, int half) {
    return (j < 4) ? (half * 8 + 2 * j) : (16 + half * 8 + 2 * (j - 4));
}

union FragH { v16h v; unsigned int u[8]; };

// ---------------------------------------------------------------------------
// f32 -> f16 copy with zero-pad of the K dimension (row-major M x Kpad)
__global__ void cvt_pad_f16_kernel(const float* __restrict__ src,
                                   _Float16* __restrict__ dst,
                                   int rows, int K, int Kpad) {
    int idx = blockIdx.x * blockDim.x + threadIdx.x;
    if (idx >= rows * Kpad) return;
    int r = idx / Kpad, k = idx - r * Kpad;
    dst[idx] = (k < K) ? (_Float16)src[r * K + k] : (_Float16)0.0f;
}

// W (K x N, f32) -> Bt (N x Kpad, f16), zero-padded; batched via blockIdx.y
__global__ void transpose_pad_f16_kernel(const float* __restrict__ W,
                                         _Float16* __restrict__ Bt,
                                         int K, int N, int Kpad) {
    long batch = blockIdx.y;
    const float* w = W + batch * (long)K * N;
    _Float16* bt = Bt + batch * (long)N * Kpad;
    int idx = blockIdx.x * blockDim.x + threadIdx.x;
    if (idx >= N * Kpad) return;
    int n = idx / Kpad, k = idx - n * Kpad;
    bt[idx] = (k < K) ? (_Float16)w[(long)k * N + n] : (_Float16)0.0f;
}

// ---------------------------------------------------------------------------
// C = relu(A @ B + bias):  A f16 row-major (M x Kpad), Bt f16 (N x Kpad).
// One wave -> one 16x16 tile; 8 waves per block. Optional f32 and f16 stores.
__global__ void wmma_gemm_bias_relu_kernel(
    const _Float16* __restrict__ A, const _Float16* __restrict__ Bt,
    const float* __restrict__ bias,
    float* __restrict__ Cf, _Float16* __restrict__ Ch,
    int N, int Kpad, int tilesN,
    long bBatchStride, long biasBatchStride,
    long cBatchStride, long cRowStride) {
    const int lane = threadIdx.x & 31;
    const int wid  = threadIdx.x >> 5;
    const long batch = blockIdx.y;
    const int tileId = blockIdx.x * 8 + wid;
    const int tileM = tileId / tilesN;
    const int tileN = tileId - tileM * tilesN;

    const int mrow = lane & 15;        // A fragment: row within tile
    const int half = lane >> 4;        // lane half selects K sub-range

    const _Float16* aRow = A + (long)(tileM * 16 + mrow) * Kpad;
    const _Float16* bRow = Bt + batch * bBatchStride
                              + (long)(tileN * 16 + mrow) * Kpad;  // col = lane&15

    v8f acc = {};
    for (int k0 = 0; k0 < Kpad; k0 += 32) {
        FragH a, b;
        #pragma unroll
        for (int j = 0; j < 8; ++j) {
            int kb = k0 + frag_kbase(j, half);
            a.u[j] = *reinterpret_cast<const unsigned int*>(aRow + kb);
            b.u[j] = *reinterpret_cast<const unsigned int*>(bRow + kb);
        }
        acc = __builtin_amdgcn_wmma_f32_16x16x32_f16(
            /*neg_a=*/false, a.v, /*neg_b=*/false, b.v,
            /*c_mod=*/(short)0, acc, /*reuse_a=*/false, /*reuse_b=*/false);
    }

    // D layout: lane holds column n = lane&15; VGPR r holds row r (+8 for hi lanes)
    const int nCol = tileN * 16 + (lane & 15);
    const float bval = bias[biasBatchStride * batch + nCol];
    #pragma unroll
    for (int r = 0; r < 8; ++r) {
        int m = tileM * 16 + r + half * 8;
        float v = acc[r] + bval;
        v = fmaxf(v, 0.0f);
        if (Cf) Cf[batch * cBatchStride + (long)m * cRowStride + nCol] = v;
        if (Ch) Ch[(long)m * N + nCol] = (_Float16)v;
    }
}

// ---------------------------------------------------------------------------
__device__ __forceinline__ float wave_reduce_add(float v) {
    #pragma unroll
    for (int off = 16; off > 0; off >>= 1) v += __shfl_xor(v, off, 32);
    return v;
}

// p[b,c] = sigmoid(dot(h[b,c,:], W_gate[c,:]) + b_gate[c]); one wave per (b,c)
__global__ void gate_kernel(const float* __restrict__ h,
                            const float* __restrict__ W_gate,
                            const float* __restrict__ b_gate,
                            float* __restrict__ p_ws,
                            float* __restrict__ p_out) {
    int waveId = blockIdx.x * 8 + (threadIdx.x >> 5);   // 0..639
    int lane = threadIdx.x & 31;
    int b = waveId / CC, c = waveId - b * CC;
    const float* hv = h + (long)b * (CC * DD) + (long)c * DD;
    const float* wv = W_gate + (long)c * DD;
    float acc = 0.0f;
    for (int d = lane; d < DD; d += 32) acc += hv[d] * wv[d];
    acc = wave_reduce_add(acc);
    if (lane == 0) {
        float p = 1.0f / (1.0f + __expf(-(acc + b_gate[c])));
        p_ws[waveId] = p;
        p_out[waveId] = p;   // p_t output (B,C,1)
    }
}

// M[b,c,k] = dot(h[b,c,:], W_mlp[:,k]); one wave per (b,c,k)
__global__ void mweight_kernel(const float* __restrict__ h,
                               const float* __restrict__ W_mlp,
                               float* __restrict__ M_ws) {
    int waveId = blockIdx.x * 8 + (threadIdx.x >> 5);   // 0..6399
    int lane = threadIdx.x & 31;
    int b = waveId / (CC * KK);
    int rem = waveId - b * (CC * KK);
    int c = rem / KK, k = rem - c * KK;
    const float* hv = h + (long)b * (CC * DD) + (long)c * DD;
    float acc = 0.0f;
    for (int d = lane; d < DD; d += 32) acc += hv[d] * W_mlp[(long)d * KK + k];
    acc = wave_reduce_add(acc);
    if (lane == 0) M_ws[waveId] = acc;                  // layout b*50 + c*10 + k
}

// ---------------------------------------------------------------------------
// Per-(b,t): gumbel-sigmoid g_t, softmax over C, 5x10 contraction, relu,
// softmax over K, transposed store (T,B,K).
__global__ void finalize_kernel(const float* __restrict__ gumbel,
                                const float* __restrict__ p_ws,
                                const float* __restrict__ M_ws,
                                const float* __restrict__ b_mlp,
                                float* __restrict__ rs_out,
                                float* __restrict__ gt_out) {
    __shared__ float sM[CC * KK];
    __shared__ float sP[CC];
    __shared__ float sB[KK];
    const int b = blockIdx.y;
    const int tid = threadIdx.x;
    if (tid < CC * KK) sM[tid] = M_ws[b * (CC * KK) + tid];
    if (tid < CC) sP[tid] = p_ws[b * CC + tid];
    if (tid < KK) sB[tid] = b_mlp[tid];
    __syncthreads();

    const int t = blockIdx.x * blockDim.x + tid;
    if (t >= TT) return;

    float g[CC];
    float mx = -3.4e38f;
    #pragma unroll
    for (int c = 0; c < CC; ++c) {
        long gi = (((long)b * CC + c) * TT + t) * 2;
        float g0 = gumbel[gi], g1 = gumbel[gi + 1];
        float z = (2.0f * sP[c] - 1.0f + g1 - g0) * TAU_INV;
        float gv = 1.0f / (1.0f + __expf(-z));          // softmax2(...)[1]
        gt_out[((long)b * CC + c) * TT + t] = gv;
        g[c] = gv;
        mx = fmaxf(mx, gv);
    }
    float s = 0.0f;
    #pragma unroll
    for (int c = 0; c < CC; ++c) { g[c] = __expf(g[c] - mx); s += g[c]; }
    float inv = 1.0f / s;

    float lk[KK];
    float mx2 = -3.4e38f;
    #pragma unroll
    for (int k = 0; k < KK; ++k) {
        float v = sB[k];
        #pragma unroll
        for (int c = 0; c < CC; ++c) v += g[c] * inv * sM[c * KK + k];
        v = fmaxf(v, 0.0f);
        lk[k] = v;
        mx2 = fmaxf(mx2, v);
    }
    float s2 = 0.0f;
    #pragma unroll
    for (int k = 0; k < KK; ++k) { lk[k] = __expf(lk[k] - mx2); s2 += lk[k]; }
    float inv2 = 1.0f / s2;
    float* o = rs_out + ((long)t * BB + b) * KK;
    #pragma unroll
    for (int k = 0; k < KK; ++k) o[k] = lk[k] * inv2;
}

// ---------------------------------------------------------------------------
extern "C" void kernel_launch(void* const* d_in, const int* in_sizes, int n_in,
                              void* d_out, int out_size, void* d_ws, size_t ws_size,
                              hipStream_t stream) {
    const float* feats   = (const float*)d_in[0];   // (128,1000)
    const float* W_adapt = (const float*)d_in[1];   // (1000,512)
    const float* b_adapt = (const float*)d_in[2];   // (512,)
    const float* W_ch    = (const float*)d_in[3];   // (5,512,512)
    const float* b_ch    = (const float*)d_in[4];   // (5,512)
    const float* W_gate  = (const float*)d_in[5];   // (5,512)
    const float* b_gate  = (const float*)d_in[6];   // (5,)
    const float* W_mlp   = (const float*)d_in[7];   // (512,10)
    const float* b_mlp   = (const float*)d_in[8];   // (10,)
    const float* gumbel  = (const float*)d_in[9];   // (128,5,1000,2)
    (void)d_in; (void)in_sizes; (void)n_in; (void)out_size; (void)ws_size;

    float* out = (float*)d_out;
    float* rs_out = out;                                   // (T,B,K) = 1,280,000
    float* gt_out = out + (long)TT * BB * KK;              // (B,C,T) =   640,000
    float* p_out  = gt_out + (long)BB * CC * TT;           // (B,C,1) =       640

    // workspace layout (byte offsets, 256-aligned)
    char* ws = (char*)d_ws;
    _Float16* feats_h  = (_Float16*)(ws + 0);                       // 128*1024
    _Float16* Wadapt_t = (_Float16*)(ws + 262144);                  // 512*1024
    _Float16* Wch_t    = (_Float16*)(ws + 1310720);                 // 5*512*512
    _Float16* rb_h     = (_Float16*)(ws + 3932160);                 // 128*512
    float*    h_f      = (float*)   (ws + 4063232);                 // 128*5*512
    float*    p_ws     = (float*)   (ws + 5373952);                 // 640
    float*    M_ws     = (float*)   (ws + 5376512);                 // 6400

    // 1) f16 staging
    cvt_pad_f16_kernel<<<(BB * KPAD1 + 255) / 256, 256, 0, stream>>>(
        feats, feats_h, BB, F_IN, KPAD1);
    transpose_pad_f16_kernel<<<dim3((DD * KPAD1 + 255) / 256, 1), 256, 0, stream>>>(
        W_adapt, Wadapt_t, F_IN, DD, KPAD1);
    transpose_pad_f16_kernel<<<dim3((DD * DD + 255) / 256, CC), 256, 0, stream>>>(
        W_ch, Wch_t, DD, DD, DD);

    // 2) GEMM1: r_b = relu(feats @ W_adapt + b_adapt), f16 out only
    //    tiles = (128/16)*(512/16) = 256 -> 32 blocks of 8 waves
    wmma_gemm_bias_relu_kernel<<<dim3(32, 1), 256, 0, stream>>>(
        feats_h, Wadapt_t, b_adapt,
        /*Cf=*/nullptr, /*Ch=*/rb_h,
        DD, KPAD1, DD / 16,
        0, 0, 0, DD);

    // 3) GEMM2 (batched over c): h[:,c,:] = relu(r_b @ W_ch[c] + b_ch[c]), f32 out
    wmma_gemm_bias_relu_kernel<<<dim3(32, CC), 256, 0, stream>>>(
        rb_h, Wch_t, b_ch,
        /*Cf=*/h_f, /*Ch=*/nullptr,
        DD, DD, DD / 16,
        (long)DD * DD, DD, DD, (long)CC * DD);

    // 4) gates p[b,c] (+ p_t output): 640 waves
    gate_kernel<<<80, 256, 0, stream>>>(h_f, W_gate, b_gate, p_ws, p_out);

    // 5) M[b,c,k] = h[b,c,:] . W_mlp[:,k]: 6400 waves
    mweight_kernel<<<800, 256, 0, stream>>>(h_f, W_mlp, M_ws);

    // 6) finalize per (b,t): g_t, v_t softmax, mlp+relu, softmax, transpose-store
    finalize_kernel<<<dim3((TT + 127) / 128, BB), 128, 0, stream>>>(
        gumbel, p_ws, M_ws, b_mlp, rs_out, gt_out);
}
